// MRC_Bert_BiLSTM_CRF_57698590655062
// MI455X (gfx1250) — matrix operand order repaired and checked by reading, not compiled, not run on hardware
//
#include <hip/hip_runtime.h>
#include <cstddef>
#include <cstdint>

// ---------------------------------------------------------------------------
// Types for CDNA5 WMMA (wave32): 16x16x32 bf16 -> f32
// ---------------------------------------------------------------------------
typedef __attribute__((ext_vector_type(16))) __bf16 v16bf;
typedef __attribute__((ext_vector_type(8)))  float  v8f;

union Frag {          // assemble a 16-element bf16 operand from two 16B loads
    uint4 u[2];
    v16bf v;
};

__device__ __forceinline__ unsigned short f32_to_bf16_bits(float f) {
    unsigned int u = __float_as_uint(f);
    u += 0x7FFFu + ((u >> 16) & 1u);          // round-to-nearest-even
    return (unsigned short)(u >> 16);
}

__device__ __forceinline__ float sigmoidf_fast(float x) {
    return 1.0f / (1.0f + __expf(-x));
}
__device__ __forceinline__ float tanhf_fast(float x) {
    // tanh(x) = 2*sigmoid(2x) - 1, via v_exp_f32
    return 2.0f / (1.0f + __expf(-2.0f * x)) - 1.0f;
}

static constexpr int B  = 64;
static constexpr int T  = 512;
static constexpr int D  = 768;
static constexpr int H  = 128;
static constexpr int G  = 4 * H;   // 512
static constexpr int KT = 11;      // tag count
static constexpr int M  = B * T;   // 32768 rows

// ---------------------------------------------------------------------------
// f32 -> bf16 (bits) elementwise convert, grid-stride
// ---------------------------------------------------------------------------
__global__ void cvt_f32_bf16(const float* __restrict__ in,
                             unsigned short* __restrict__ out, size_t n) {
    size_t i = (size_t)blockIdx.x * blockDim.x + threadIdx.x;
    size_t stride = (size_t)gridDim.x * blockDim.x;
    for (; i < n; i += stride) out[i] = f32_to_bf16_bits(in[i]);
}

// ---------------------------------------------------------------------------
// xp GEMM:  C[M x 512] = A[M x Kd](bf16) * W[512 x Kd]^T (bf16) + bias
// Block = 256 threads = 8 waves, one 16-row M-tile per block; wave w covers
// N in [64w, 64w+64) with 4 WMMA accumulators. K-loop in steps of 32,
// manually double-buffered so WMMAs of step k overlap loads of step k+32.
// ---------------------------------------------------------------------------
__global__ __launch_bounds__(256)
void gemm_xp(const unsigned short* __restrict__ A,
             const unsigned short* __restrict__ W,
             const float* __restrict__ bias,
             float* __restrict__ C, int Kd) {
    const int lane = threadIdx.x & 31;
    const int wave = threadIdx.x >> 5;
    const int hfl  = lane >> 4;      // lane half (0/1)
    const int l15  = lane & 15;
    const int m0   = blockIdx.x * 16;

    v8f z = {0.f,0.f,0.f,0.f,0.f,0.f,0.f,0.f};
    v8f acc[4] = {z, z, z, z};

    const unsigned short* arow = A + (size_t)(m0 + l15) * Kd;
    const unsigned short* wbase[4];
#pragma unroll
    for (int tN = 0; tN < 4; ++tN) {
        const int n = wave * 64 + tN * 16 + l15;
        wbase[tN] = W + (size_t)n * Kd + 8 * hfl;
    }

    // stage 0 preload (k0 = 0)
    Frag a0, b0[4];
    a0.u[0] = *(const uint4*)(arow + 8 * hfl);
    a0.u[1] = *(const uint4*)(arow + 16 + 8 * hfl);
#pragma unroll
    for (int tN = 0; tN < 4; ++tN) {
        b0[tN].u[0] = ((const uint4*)wbase[tN])[0];
        b0[tN].u[1] = ((const uint4*)wbase[tN])[2];
    }

    int k0 = 0;
    for (; k0 + 32 < Kd; k0 += 32) {
        const int kn = k0 + 32;
        // prefetch + load next-stage fragments first ...
        Frag a1, b1[4];
        a1.u[0] = *(const uint4*)(arow + kn + 8 * hfl);
        a1.u[1] = *(const uint4*)(arow + kn + 16 + 8 * hfl);
        __builtin_prefetch(arow + kn + 32, 0, 1);   // global_prefetch ahead
#pragma unroll
        for (int tN = 0; tN < 4; ++tN) {
            b1[tN].u[0] = ((const uint4*)(wbase[tN] + kn))[0];
            b1[tN].u[1] = ((const uint4*)(wbase[tN] + kn))[2];
        }
        // ... then issue WMMAs for the current stage (overlaps the loads)
#pragma unroll
        for (int tN = 0; tN < 4; ++tN) {
            acc[tN] = __builtin_amdgcn_wmma_f32_16x16x32_bf16(
                false, a0.v, false, b0[tN].v, (short)0, acc[tN], false, false);
        }
        a0 = a1;
#pragma unroll
        for (int tN = 0; tN < 4; ++tN) b0[tN] = b1[tN];
    }
    // epilogue stage
#pragma unroll
    for (int tN = 0; tN < 4; ++tN) {
        acc[tN] = __builtin_amdgcn_wmma_f32_16x16x32_bf16(
            false, a0.v, false, b0[tN].v, (short)0, acc[tN], false, false);
    }

#pragma unroll
    for (int tN = 0; tN < 4; ++tN) {
        const int n  = wave * 64 + tN * 16 + l15;
        const float bn = bias[n];
#pragma unroll
        for (int r = 0; r < 8; ++r) {
            const int m = m0 + r + 8 * hfl;
            C[(size_t)m * G + n] = acc[tN][r] + bn;
        }
    }
}

// ---------------------------------------------------------------------------
// Persistent LSTM scan. grid = (4 batch-groups of 16, 2 directions).
// h (bf16) and c (f32) live in LDS. Whh is loop-invariant: each wave hoists
// its 16 B-fragments (4 N-tiles x 4 K-steps, 128 VGPRs) into registers before
// the T loop, so the latency-critical step is only ds_load_b128 + WMMA +
// the xp gather (prefetched one step ahead).
// ---------------------------------------------------------------------------
__global__ __launch_bounds__(256, 1)
void lstm_layer(const float* __restrict__ xp_f, const float* __restrict__ xp_b,
                const unsigned short* __restrict__ Whh_f,
                const unsigned short* __restrict__ Whh_b,
                unsigned short* __restrict__ out_bf16,
                float* __restrict__ out_f32) {
    __shared__ __align__(16) unsigned short hsm[16][H];  // 4 KB
    __shared__ float csm[16][H];                         // 8 KB
    __shared__ __align__(16) float gsm[16][G];           // 32 KB

    const int dir   = blockIdx.y;
    const int bbase = blockIdx.x * 16;
    const float*          xp  = dir ? xp_b  : xp_f;
    const unsigned short* Whh = dir ? Whh_b : Whh_f;

    const int lane = threadIdx.x & 31;
    const int wave = threadIdx.x >> 5;
    const int hfl  = lane >> 4;
    const int l15  = lane & 15;

    // hoist Whh B-fragments into registers (invariant over all T steps)
    Frag bw[4][4];
#pragma unroll
    for (int tN = 0; tN < 4; ++tN) {
        const int n = wave * 64 + tN * 16 + l15;
#pragma unroll
        for (int kt = 0; kt < 4; ++kt) {
            const unsigned short* wr = Whh + (size_t)n * H + kt * 32 + 8 * hfl;
            bw[tN][kt].u[0] = ((const uint4*)wr)[0];
            bw[tN][kt].u[1] = ((const uint4*)wr)[2];
        }
    }

    for (int e = threadIdx.x; e < 16 * H; e += 256) {
        hsm[e >> 7][e & 127] = 0;
        csm[e >> 7][e & 127] = 0.0f;
    }
    __syncthreads();

    for (int s = 0; s < T; ++s) {
        const int t = dir ? (T - 1 - s) : s;

        v8f z = {0.f,0.f,0.f,0.f,0.f,0.f,0.f,0.f};
        v8f acc[4] = {z, z, z, z};
#pragma unroll
        for (int kt = 0; kt < 4; ++kt) {
            const int k0 = kt * 32;
            Frag a;   // h fragment from LDS (ds_load_b128 x2)
            a.u[0] = *(const uint4*)&hsm[l15][k0 + 8 * hfl];
            a.u[1] = *(const uint4*)&hsm[l15][k0 + 16 + 8 * hfl];
#pragma unroll
            for (int tN = 0; tN < 4; ++tN) {
                acc[tN] = __builtin_amdgcn_wmma_f32_16x16x32_bf16(
                    false, a.v, false, bw[tN][kt].v, (short)0, acc[tN],
                    false, false);
            }
        }

        // prefetch next timestep's xp rows (16 rows x 2KB, one line/thread)
        if (s + 1 < T) {
            const int tn = dir ? (t - 1) : (t + 1);
            const int e  = threadIdx.x;               // 256 threads = 16x16
            const int br = bbase + (e >> 4);
            const int ch = (e & 15) * 32;
            __builtin_prefetch(&xp[((size_t)br * T + tn) * G + ch], 0, 1);
        }

        // g = acc + xp[b, t, :]
#pragma unroll
        for (int tN = 0; tN < 4; ++tN) {
            const int n = wave * 64 + tN * 16 + l15;
#pragma unroll
            for (int r = 0; r < 8; ++r) {
                const int ml = r + 8 * hfl;
                const int bg = bbase + ml;
                gsm[ml][n] = acc[tN][r] + xp[((size_t)bg * T + t) * G + n];
            }
        }
        __syncthreads();

        // gates: i | f | g | o  (layout matches jnp.split(g, 4))
        for (int e = threadIdx.x; e < 16 * H; e += 256) {
            const int ml = e >> 7, hh = e & 127;
            const float gi = gsm[ml][hh];
            const float gf = gsm[ml][H + hh];
            const float gg = gsm[ml][2 * H + hh];
            const float go = gsm[ml][3 * H + hh];
            const float cn = sigmoidf_fast(gf) * csm[ml][hh] +
                             sigmoidf_fast(gi) * tanhf_fast(gg);
            const float hn = sigmoidf_fast(go) * tanhf_fast(cn);
            csm[ml][hh] = cn;
            hsm[ml][hh] = f32_to_bf16_bits(hn);
            const size_t o = ((size_t)(bbase + ml) * T + t) * (2 * H) +
                             (size_t)dir * H + hh;
            out_bf16[o] = f32_to_bf16_bits(hn);
            out_f32[o]  = hn;
        }
        __syncthreads();
    }
}

// ---------------------------------------------------------------------------
// feats[m][k] = enc1[m][:256] . Wlin[k][:256] + blin[k]   (K=11, VALU f32)
// ---------------------------------------------------------------------------
__global__ __launch_bounds__(256)
void linear_feats(const float* __restrict__ enc,
                  const float* __restrict__ Wlin,
                  const float* __restrict__ blin,
                  float* __restrict__ feats) {
    const int m = blockIdx.x * blockDim.x + threadIdx.x;
    if (m >= M) return;
    const float* row = enc + (size_t)m * (2 * H);
#pragma unroll 1
    for (int k = 0; k < KT; ++k) {
        const float* w = Wlin + (size_t)k * (2 * H);
        float s = blin[k];
        for (int j = 0; j < 2 * H; ++j) s = fmaf(row[j], w[j], s);
        feats[(size_t)m * KT + k] = s;
    }
}

// ---------------------------------------------------------------------------
// CRF numerator (per-batch gold-path score)
// ---------------------------------------------------------------------------
__global__ void crf_num(const float* __restrict__ feats,
                        const int* __restrict__ tags,
                        const int* __restrict__ mask,
                        const float* __restrict__ start,
                        const float* __restrict__ endv,
                        const float* __restrict__ trans,
                        float* __restrict__ num) {
    const int b = threadIdx.x;                    // 64 threads
    const int*   tg = tags + (size_t)b * T;
    const int*   mk = mask + (size_t)b * T;
    const float* em = feats + (size_t)b * T * KT;
    int prev = tg[0];
    float acc = start[prev] + em[prev];
    int cnt = 1;
    for (int t = 1; t < T; ++t) {
        const int tt = tg[t];
        if (mk[t]) {
            acc += trans[prev * KT + tt] + em[(size_t)t * KT + tt];
        }
        prev = tt;
        cnt += (mk[t] != 0);
    }
    int total = (mk[0] != 0) ? cnt : (cnt - 1);
    if (total < 1) total = 1;
    acc += endv[tg[total - 1]];
    num[b] = acc;
}

// ---------------------------------------------------------------------------
// CRF forward algorithm (partition function). One block of 1024 threads:
// thread -> (b = tid/16, j = tid%16), active when j < 11. Scores in LDS.
// ---------------------------------------------------------------------------
__global__ __launch_bounds__(1024)
void crf_fwd(const float* __restrict__ feats,
             const int* __restrict__ mask,
             const float* __restrict__ start,
             const float* __restrict__ endv,
             const float* __restrict__ trans,
             float* __restrict__ den) {
    __shared__ float sc[B][16];
    __shared__ float tr[16][16];
    const int tid = threadIdx.x;
    const int b = tid >> 4;
    const int j = tid & 15;
    const bool active = (j < KT);

    if (tid < 256) {
        const int r = tid >> 4, c = tid & 15;
        tr[r][c] = (r < KT && c < KT) ? trans[r * KT + c] : 0.0f;
    }
    if (active) sc[b][j] = start[j] + feats[(size_t)b * T * KT + j];
    __syncthreads();

    for (int t = 1; t < T; ++t) {
        float nxt = 0.0f;
        if (active) {
            float mx = -3.4e38f;
#pragma unroll
            for (int i = 0; i < KT; ++i)
                mx = fmaxf(mx, sc[b][i] + tr[i][j]);
            float s = 0.0f;
#pragma unroll
            for (int i = 0; i < KT; ++i)
                s += __expf(sc[b][i] + tr[i][j] - mx);
            nxt = mx + __logf(s) + feats[((size_t)b * T + t) * KT + j];
        }
        __syncthreads();
        if (active && mask[(size_t)b * T + t]) sc[b][j] = nxt;
        __syncthreads();
    }

    if (j == 0) {
        float mx = -3.4e38f;
#pragma unroll
        for (int k = 0; k < KT; ++k) mx = fmaxf(mx, sc[b][k] + endv[k]);
        float s = 0.0f;
#pragma unroll
        for (int k = 0; k < KT; ++k) s += __expf(sc[b][k] + endv[k] - mx);
        den[b] = mx + __logf(s);
    }
}

__global__ void crf_final(const float* __restrict__ num,
                          const float* __restrict__ den,
                          float* __restrict__ out) {
    __shared__ float red[B];
    const int b = threadIdx.x;
    red[b] = den[b] - num[b];
    __syncthreads();
    if (b == 0) {
        float s = 0.0f;
        for (int i = 0; i < B; ++i) s += red[i];
        out[0] = s / (float)B;
    }
}

// ---------------------------------------------------------------------------
// Host launch
// ---------------------------------------------------------------------------
extern "C" void kernel_launch(void* const* d_in, const int* in_sizes, int n_in,
                              void* d_out, int out_size, void* d_ws, size_t ws_size,
                              hipStream_t stream) {
    const float* embeds = (const float*)d_in[0];
    const int*   tags   = (const int*)d_in[1];
    const int*   amask  = (const int*)d_in[2];
    const float* Wih0f  = (const float*)d_in[3];
    const float* Whh0f  = (const float*)d_in[4];
    const float* b0f    = (const float*)d_in[5];
    const float* Wih0b  = (const float*)d_in[6];
    const float* Whh0b  = (const float*)d_in[7];
    const float* b0b    = (const float*)d_in[8];
    const float* Wih1f  = (const float*)d_in[9];
    const float* Whh1f  = (const float*)d_in[10];
    const float* b1f    = (const float*)d_in[11];
    const float* Wih1b  = (const float*)d_in[12];
    const float* Whh1b  = (const float*)d_in[13];
    const float* b1b    = (const float*)d_in[14];
    const float* Wlin   = (const float*)d_in[15];
    const float* blin   = (const float*)d_in[16];
    const float* startv = (const float*)d_in[17];
    const float* endv   = (const float*)d_in[18];
    const float* trans  = (const float*)d_in[19];

    char* ws = (char*)d_ws;
    size_t off = 0;
    auto alloc = [&](size_t bytes) -> size_t {
        size_t o = off;
        off = (off + bytes + 255) & ~(size_t)255;
        return o;
    };

    const size_t off_eb    = alloc((size_t)M * D * 2);       // embeds bf16 (reused as enc1 f32)
    const size_t off_w0f   = alloc((size_t)G * D * 2);
    const size_t off_w0b   = alloc((size_t)G * D * 2);
    const size_t off_u0f   = alloc((size_t)G * H * 2);
    const size_t off_u0b   = alloc((size_t)G * H * 2);
    const size_t off_w1f   = alloc((size_t)G * 2 * H * 2);
    const size_t off_w1b   = alloc((size_t)G * 2 * H * 2);
    const size_t off_u1f   = alloc((size_t)G * H * 2);
    const size_t off_u1b   = alloc((size_t)G * H * 2);
    const size_t off_xpf   = alloc((size_t)M * G * 4);
    const size_t off_xpb   = alloc((size_t)M * G * 4);
    const size_t off_enc0b = alloc((size_t)M * 2 * H * 2);   // layer0 out bf16
    const size_t off_dump  = alloc((size_t)M * 2 * H * 4);   // shared dump buffer
    const size_t off_feats = alloc((size_t)M * KT * 4);
    const size_t off_num   = alloc(B * 4);
    const size_t off_den   = alloc(B * 4);

    unsigned short* eb    = (unsigned short*)(ws + off_eb);
    unsigned short* w0f   = (unsigned short*)(ws + off_w0f);
    unsigned short* w0b   = (unsigned short*)(ws + off_w0b);
    unsigned short* u0f   = (unsigned short*)(ws + off_u0f);
    unsigned short* u0b   = (unsigned short*)(ws + off_u0b);
    unsigned short* w1f   = (unsigned short*)(ws + off_w1f);
    unsigned short* w1b   = (unsigned short*)(ws + off_w1b);
    unsigned short* u1f   = (unsigned short*)(ws + off_u1f);
    unsigned short* u1b   = (unsigned short*)(ws + off_u1b);
    float*          xpf   = (float*)(ws + off_xpf);
    float*          xpb   = (float*)(ws + off_xpb);
    unsigned short* enc0b = (unsigned short*)(ws + off_enc0b);
    float*          dumpf = (float*)(ws + off_dump);          // layer0 f32 dump
    unsigned short* dumpb = (unsigned short*)(ws + off_dump); // layer1 bf16 dump
    float*          enc1  = (float*)(ws + off_eb);            // reuse embeds-bf16 region
    float*          feats = (float*)(ws + off_feats);
    float*          numb  = (float*)(ws + off_num);
    float*          denb  = (float*)(ws + off_den);

    // 1) converts
    struct Cvt { const float* s; unsigned short* d; size_t n; };
    const Cvt cv[] = {
        {embeds, eb,  (size_t)M * D},
        {Wih0f,  w0f, (size_t)G * D},  {Wih0b, w0b, (size_t)G * D},
        {Whh0f,  u0f, (size_t)G * H},  {Whh0b, u0b, (size_t)G * H},
        {Wih1f,  w1f, (size_t)G * 2 * H}, {Wih1b, w1b, (size_t)G * 2 * H},
        {Whh1f,  u1f, (size_t)G * H},  {Whh1b, u1b, (size_t)G * H},
    };
    for (int i = 0; i < 9; ++i) {
        int blocks = (int)((cv[i].n + 1023) / 1024);
        if (blocks > 4096) blocks = 4096;
        cvt_f32_bf16<<<blocks, 256, 0, stream>>>(cv[i].s, cv[i].d, cv[i].n);
    }

    // 2) layer-0 input projections (WMMA)
    gemm_xp<<<M / 16, 256, 0, stream>>>(eb, w0f, b0f, xpf, D);
    gemm_xp<<<M / 16, 256, 0, stream>>>(eb, w0b, b0b, xpb, D);

    // 3) layer-0 recurrent scan (4 batch groups x 2 directions)
    lstm_layer<<<dim3(4, 2), 256, 0, stream>>>(xpf, xpb, u0f, u0b, enc0b, dumpf);

    // 4) layer-1 input projections (reuse xp buffers)
    gemm_xp<<<M / 16, 256, 0, stream>>>(enc0b, w1f, b1f, xpf, 2 * H);
    gemm_xp<<<M / 16, 256, 0, stream>>>(enc0b, w1b, b1b, xpb, 2 * H);

    // 5) layer-1 recurrent scan -> enc1 (f32)
    lstm_layer<<<dim3(4, 2), 256, 0, stream>>>(xpf, xpb, u1f, u1b, dumpb, enc1);

    // 6) emissions
    linear_feats<<<M / 256, 256, 0, stream>>>(enc1, Wlin, blin, feats);

    // 7) CRF
    crf_num<<<1, B, 0, stream>>>(feats, tags, amask, startv, endv, trans, numb);
    crf_fwd<<<1, 1024, 0, stream>>>(feats, amask, startv, endv, trans, denb);
    crf_final<<<1, B, 0, stream>>>(numb, denb, (float*)d_out);
}
